// HomogeneousDelayedHORN_71889162600900
// MI455X (gfx1250) — matrix operand order repaired and checked by reading, not compiled, not run on hardware
//
#include <hip/hip_runtime.h>
#include <math.h>

// ---------------------------------------------------------------------------
// HomogeneousDelayedHORN on MI455X (gfx1250)
//   - 64 persistent workgroups: 8 batch-groups (16 rows) x 8 col-slices (64)
//   - W_hh slice resident in VGPRs as bf16 WMMA B-tiles (loaded once)
//   - v_wmma_f32_16x16x32_bf16; sched_group_barrier-pinned distance-3
//     software pipeline of A-tile ds_loads
//   - Tensor Data Mover: double-buffered tensor_load_to_lds stages y(t-5)
//     one step ahead (DMA overlaps WMMA); tensor_store_from_lds publishes
//     the y_new 16x64 tile into the L2 ring
//   - DELAY=5 => inter-WG sync only every 5 steps (157 phases)
// ---------------------------------------------------------------------------

typedef __attribute__((ext_vector_type(16))) __bf16 v16bf;
typedef __attribute__((ext_vector_type(8)))  __bf16 v8bf;
typedef __attribute__((ext_vector_type(8)))  float  v8f;
typedef __attribute__((ext_vector_type(4)))  unsigned int u32x4;
typedef __attribute__((ext_vector_type(8)))  int i32x8;
typedef __attribute__((ext_vector_type(4)))  int i32x4;

#define T_STEPS 784
#define BATCH   128
#define NHID    512
#define NOUT    10
#define RINGSZ  16          // power of two, > max producer/consumer skew (9)
#define NPHASE  157         // ceil(784 / 5)
#define NSLICE  8
#define TPB     128         // 4 wave32 per WG

#if defined(__AMDGCN__) && __has_builtin(__builtin_amdgcn_tensor_load_to_lds)
#define HAS_TDM 1
#else
#define HAS_TDM 0
#endif

#if defined(__AMDGCN__) && __has_builtin(__builtin_amdgcn_sched_group_barrier)
#define SGB(mask, n, id) __builtin_amdgcn_sched_group_barrier(mask, n, id)
#else
#define SGB(mask, n, id)
#endif

static __device__ __forceinline__ float dev_tanh(float v) {
#if __has_builtin(__builtin_amdgcn_tanhf)
  return __builtin_amdgcn_tanhf(v);
#else
  return tanhf(v);
#endif
}

static __device__ __forceinline__ void wait_tensor0() {
#if defined(__AMDGCN__) && __has_builtin(__builtin_amdgcn_s_wait_tensorcnt)
  __builtin_amdgcn_s_wait_tensorcnt(0);
#elif defined(__AMDGCN__)
  asm volatile("s_wait_tensorcnt 0x0" ::: "memory");
#endif
}

#if HAS_TDM
// 1D DMA: `units` x 8 bytes, global -> LDS.  D# group0/group1 per ISA 8.3/8.4.
static __device__ __forceinline__ void tdm_load_1d_8B(unsigned lds_byte,
                                                      unsigned long long ga,
                                                      unsigned units) {
  u32x4 g0;
  g0[0] = 1u;                                               // count=1
  g0[1] = lds_byte;                                         // lds_addr
  g0[2] = (unsigned)ga;                                     // global_addr lo
  g0[3] = ((unsigned)(ga >> 32) & 0x01FFFFFFu) | 0x80000000u; // hi | type=2
  i32x8 g1;
  g1[0] = 0x00030000;              // data_size=3 (8B), mask=0
  g1[1] = (int)(units << 16);      // tensor_dim0 (lo16 at bits 63:48)
  g1[2] = (int)(1u << 16);         // tensor_dim1 = 1
  g1[3] = (int)(units << 16);      // tile_dim0
  g1[4] = 0;                       // tile_dim1/2 unused
  g1[5] = (int)units;              // tensor_dim0_stride
  g1[6] = 0;
  g1[7] = 0;
  i32x4 z4 = {0, 0, 0, 0};
  i32x8 z8 = {0, 0, 0, 0, 0, 0, 0, 0};
  __builtin_amdgcn_tensor_load_to_lds(g0, g1, z4, z4, z8, 16 /*SCOPE_DEV*/);
}

// 2D DMA: h rows of w_units x 8 bytes, LDS (contiguous) -> global (strided).
static __device__ __forceinline__ void tdm_store_2d_8B(unsigned lds_byte,
                                                       unsigned long long ga,
                                                       unsigned w_units,
                                                       unsigned h,
                                                       unsigned stride_units) {
  u32x4 g0;
  g0[0] = 1u;
  g0[1] = lds_byte;
  g0[2] = (unsigned)ga;
  g0[3] = ((unsigned)(ga >> 32) & 0x01FFFFFFu) | 0x80000000u;
  i32x8 g1;
  g1[0] = 0x00030000;              // data_size=3 (8B)
  g1[1] = (int)(w_units << 16);    // tensor_dim0
  g1[2] = (int)(h << 16);          // tensor_dim1
  g1[3] = (int)(w_units << 16);    // tile_dim0
  g1[4] = (int)h;                  // tile_dim1
  g1[5] = (int)stride_units;       // tensor_dim0_stride (row pitch)
  g1[6] = 0;
  g1[7] = 0;
  i32x4 z4 = {0, 0, 0, 0};
  i32x8 z8 = {0, 0, 0, 0, 0, 0, 0, 0};
  __builtin_amdgcn_tensor_store_from_lds(g0, g1, z4, z4, z8, 16 /*SCOPE_DEV*/);
}
#endif

__global__ void __launch_bounds__(64)
horn_init(int* progress) {
  if (threadIdx.x < 64) progress[threadIdx.x] = 0;
}

__global__ void __launch_bounds__(TPB, 1)
horn_main(const float* __restrict__ seq,     // [128, 784]
          const float* __restrict__ w_ih,    // [512]
          const float* __restrict__ b_ih,    // [512]
          const float* __restrict__ w_hh,    // [512, 512] row n = out col
          const float* __restrict__ b_hh,    // [512]
          const float* __restrict__ w_out,   // [10, 512]
          __bf16* ring, int* progress, float* partial)
{
  __shared__ __align__(16) __bf16 ldsY[2][16 * NHID];  // double-buffered slab
  __shared__ __align__(16) __bf16 yLDS[16 * 64];       // staged y_new tile
  __shared__ float sLDS[16];
  __shared__ float xLDS[16 * 64];

  const int wg   = blockIdx.x;
  const int g    = wg >> 3;          // batch group   (rows r0..r0+15)
  const int r0   = g << 4;
  const int c0   = (wg & 7) << 6;    // column slice  (cols c0..c0+63)
  const int tid  = threadIdx.x;
  const int lane = tid & 31;
  const int wv   = tid >> 5;         // wave id, owns 16 cols
  const int hi   = lane >> 4;        // lane half (WMMA layout)
  const int ln   = lane & 15;
  const int colbase = c0 + (wv << 4);
  const int mycol   = colbase + ln;

#if HAS_TDM
  const unsigned ldsY_off0 = (unsigned)(unsigned long long)(uintptr_t)&ldsY[0][0];
  const unsigned ldsY_off1 = (unsigned)(unsigned long long)(uintptr_t)&ldsY[1][0];
  const unsigned yLDS_off  = (unsigned)(unsigned long long)(uintptr_t)&yLDS[0];
#endif

  // ---- W_hh slice -> resident bf16 B-tiles (B layout: lane = N, elem = K) --
  v16bf Bt[16];
#pragma unroll
  for (int ks = 0; ks < 16; ++ks) {
#pragma unroll
    for (int e = 0; e < 16; ++e) {
      Bt[ks][e] = (__bf16)w_hh[(size_t)(colbase + ln) * NHID + ks * 32 + 8 * hi + e + (e & 8)];
    }
  }

  const float wih  = w_ih[mycol];
  const float bias = b_ih[mycol] + b_hh[mycol];
  const float inv_sqrt_n = 0.04419417382415922f;   // 1/sqrt(512)

  // per-lane state: (row = r0 + j + 8*hi, col = mycol), j = 0..7  (C layout)
  float xs[8], ys[8];
#pragma unroll
  for (int j = 0; j < 8; ++j) { xs[j] = 0.f; ys[j] = 0.f; }

  for (int p = 0; p < NPHASE; ++p) {
    // wait until all 8 slice-WGs of this batch group finished phase p-1
    if (p > 0 && tid < NSLICE) {
      const int peer = (g << 3) + tid;
      while (__hip_atomic_load(&progress[peer], __ATOMIC_ACQUIRE,
                               __HIP_MEMORY_SCOPE_AGENT) < p) {
        __builtin_amdgcn_s_sleep(2);
      }
    }
    __syncthreads();

    const int tbeg = 5 * p;
    int tend = tbeg + 5; if (tend > T_STEPS) tend = T_STEPS;

#if HAS_TDM
    // phase prologue: kick off the DMA for the first step of this phase
    if (wv == 0 && tbeg >= 5) {
      const __bf16* src =
          ring + ((size_t)((tbeg - 5) & (RINGSZ - 1)) * BATCH + r0) * NHID;
      tdm_load_1d_8B((tbeg & 1) ? ldsY_off1 : ldsY_off0,
                     (unsigned long long)(uintptr_t)src, 2048u);
    }
#endif

    for (int t = tbeg; t < tend; ++t) {
#if HAS_TDM
      if (wv == 0) {
        wait_tensor0();   // load(t) landed in ldsY[t&1]; store(t-1) drained
        // issue next step's DMA now so it overlaps this step's WMMA work
        if (t + 1 < tend && t + 1 >= 5) {
          const __bf16* src =
              ring + ((size_t)((t - 4) & (RINGSZ - 1)) * BATCH + r0) * NHID;
          tdm_load_1d_8B(((t + 1) & 1) ? ldsY_off1 : ldsY_off0,
                         (unsigned long long)(uintptr_t)src, 2048u);
        }
      }
#else
      __syncthreads();
      if (t >= 5) {
        const u32x4* src = (const u32x4*)
            (ring + ((size_t)((t - 5) & (RINGSZ - 1)) * BATCH + r0) * NHID);
        u32x4* dst = (u32x4*)&ldsY[t & 1][0];
#pragma unroll
        for (int i = 0; i < 8; ++i) dst[tid + TPB * i] = src[tid + TPB * i];
      }
#endif
      if (tid < 16) sLDS[tid] = seq[(size_t)(r0 + tid) * T_STEPS + t];
      __syncthreads();

      v8f acc = {0.f, 0.f, 0.f, 0.f, 0.f, 0.f, 0.f, 0.f};
      if (t >= 5) {
        // A layout (16x32 bf16): lane ln = row, two 8-elem K chunks at
        // K = ks*32 + 8*hi and K = ks*32 + 16 + 8*hi
        const __bf16* arow = &ldsY[t & 1][ln * NHID + 8 * hi];
        auto LOADA = [&](int ks) -> v16bf {
          const v8bf* cp = (const v8bf*)(arow + ks * 32);
          v8bf a0 = cp[0];
          v8bf a1 = cp[2];                  // +32 B = +16 elements
          return __builtin_shufflevector(a0, a1,
              0,1,2,3,4,5,6,7,8,9,10,11,12,13,14,15);
        };
        // distance-3 software pipeline over the 16 K-tiles
        v16bf Ab[4];
        Ab[0] = LOADA(0);
        Ab[1] = LOADA(1);
        Ab[2] = LOADA(2);
#pragma unroll
        for (int ks = 0; ks < 16; ++ks) {
          if (ks + 3 < 16) Ab[(ks + 3) & 3] = LOADA(ks + 3);
          acc = __builtin_amdgcn_wmma_f32_16x16x32_bf16(
              false, Ab[ks & 3], false, Bt[ks], (short)0, acc, false, false);
        }
        // pin the schedule: loads run 3 tiles ahead of the WMMA chain
        SGB(0x100, 6, 0);                      // 6 ds-reads (tiles 0..2)
#pragma unroll
        for (int i = 0; i < 13; ++i) {
          SGB(0x008, 1, 0);                    // 1 WMMA
          SGB(0x100, 2, 0);                    // 2 ds-reads (tile i+3)
        }
        SGB(0x008, 3, 0);                      // tail WMMAs
      }

      // oscillator update; stage y_new tile for the TDM store
#if !HAS_TDM
      __bf16* yslab = ring + (size_t)(t & (RINGSZ - 1)) * BATCH * NHID;
#endif
#pragma unroll
      for (int j = 0; j < 8; ++j) {
        const float s     = sLDS[j + 8 * hi];
        const float a_in  = (fmaf(s, wih, acc[j]) + bias) * inv_sqrt_n;
        const float accel = dev_tanh(a_in) - 0.2f * ys[j] - xs[j];
        const float xn = xs[j] + ys[j];     // H = 1, OMEGA = 1, GAMMA = 0.1
        const float yn = ys[j] + accel;
        xs[j] = xn;
        ys[j] = yn;
#if HAS_TDM
        yLDS[(j + 8 * hi) * 64 + (wv << 4) + ln] = (__bf16)yn;
#else
        yslab[(size_t)(r0 + j + 8 * hi) * NHID + mycol] = (__bf16)yn;
#endif
      }
      __syncthreads();
#if HAS_TDM
      // wave0: DMA y_new 16x64 tile LDS -> ring slab t%16 (row pitch 512)
      if (wv == 0) {
        const __bf16* dst = ring + (size_t)(t & (RINGSZ - 1)) * BATCH * NHID +
                            (size_t)r0 * NHID + c0;
        tdm_store_2d_8B(yLDS_off, (unsigned long long)(uintptr_t)dst,
                        16u /* 64 bf16 = 16x8B */, 16u /* rows */,
                        128u /* 512 bf16 pitch = 128x8B */);
      }
#endif
    }

    if (wv == 0) wait_tensor0();   // all slab stores of this phase complete
    __syncthreads();
    if (tid == 0) {
      __threadfence();
      __hip_atomic_store(&progress[wg], p + 1, __ATOMIC_RELEASE,
                         __HIP_MEMORY_SCOPE_AGENT);
    }
  }

  // ---- readout partials: x_fin[16 x 64] . w_out[:, c0:c0+64]^T ------------
#pragma unroll
  for (int j = 0; j < 8; ++j)
    xLDS[(j + 8 * hi) * 64 + (wv << 4) + ln] = xs[j];
  __syncthreads();
  for (int idx = tid; idx < 16 * NOUT; idx += TPB) {
    const int r = idx / NOUT;
    const int o = idx - r * NOUT;
    float sum = 0.f;
    for (int c = 0; c < 64; ++c)
      sum += xLDS[r * 64 + c] * w_out[(size_t)o * NHID + c0 + c];
    partial[((size_t)wg * 16 + r) * NOUT + o] = sum;
  }
}

__global__ void __launch_bounds__(256)
horn_final(const float* __restrict__ partial,
           const float* __restrict__ b_out,
           float* __restrict__ out) {
  const int i = blockIdx.x * 256 + threadIdx.x;
  if (i >= BATCH * NOUT) return;
  const int b  = i / NOUT;
  const int o  = i - b * NOUT;
  const int gg = b >> 4;
  const int r  = b & 15;
  float sum = b_out[o];
  for (int s = 0; s < NSLICE; ++s)
    sum += partial[((size_t)((gg << 3) + s) * 16 + r) * NOUT + o];
  out[i] = sum;
}

extern "C" void kernel_launch(void* const* d_in, const int* in_sizes, int n_in,
                              void* d_out, int out_size, void* d_ws, size_t ws_size,
                              hipStream_t stream) {
  (void)in_sizes; (void)n_in; (void)out_size; (void)ws_size;
  const float* seq   = (const float*)d_in[0];
  const float* w_ih  = (const float*)d_in[1];
  const float* b_ih  = (const float*)d_in[2];
  const float* w_hh  = (const float*)d_in[3];
  const float* b_hh  = (const float*)d_in[4];
  const float* w_out = (const float*)d_in[5];
  const float* b_out = (const float*)d_in[6];
  float* out = (float*)d_out;

  // workspace layout: [ring 2 MB bf16][progress 256 B][partials 40 KB]
  char* ws = (char*)d_ws;
  const size_t ring_bytes = (size_t)RINGSZ * BATCH * NHID * 2;
  __bf16* ring = (__bf16*)ws;
  int*    prog = (int*)(ws + ring_bytes);
  float*  part = (float*)(ws + ring_bytes + 256);

  horn_init<<<1, 64, 0, stream>>>(prog);
  horn_main<<<64, TPB, 0, stream>>>(seq, w_ih, b_ih, w_hh, b_hh, w_out,
                                    ring, prog, part);
  horn_final<<<(BATCH * NOUT + 255) / 256, 256, 0, stream>>>(part, b_out, out);
}
